// S4Block_11476152615196
// MI455X (gfx1250) — matrix operand order repaired and checked by reading, not compiled, not run on hardware
//
#include <hip/hip_runtime.h>
#include <hip/hip_bf16.h>

#define BATCH   8
#define SEQ     4096
#define DMODEL  1024
#define DSTATE  16
#define DCONV   4
#define ROWS    (BATCH * SEQ)          // 32768
#define NCHUNK  16
#define CHUNKL  (SEQ / NCHUNK)         // 256

typedef __attribute__((ext_vector_type(2)))  float   v2f;
typedef __attribute__((ext_vector_type(8)))  float   v8f;
typedef __attribute__((ext_vector_type(16))) __bf16  v16bf;
typedef __attribute__((ext_vector_type(4)))  int     v4i;

// ---- CDNA5 async global->LDS path (guarded; falls back to sync staging) ----
#ifndef TRY_ASYNC
#define TRY_ASYNC 1
#endif
#if TRY_ASYNC && __has_builtin(__builtin_amdgcn_global_load_async_to_lds_b128)
#define USE_ASYNC 1
#else
#define USE_ASYNC 0
#endif

__device__ __forceinline__ void async_ld16(const void* g, void* l) {
#if USE_ASYNC
    __builtin_amdgcn_global_load_async_to_lds_b128(
        (__attribute__((address_space(1))) v4i*)(void*)g,
        (__attribute__((address_space(3))) v4i*)l, 0, 0);
#else
    (void)g; (void)l;
#endif
}

__device__ __forceinline__ void wait_async0() {
#if USE_ASYNC
#if __has_builtin(__builtin_amdgcn_s_wait_asynccnt)
    __builtin_amdgcn_s_wait_asynccnt(0);
#else
    asm volatile("s_wait_asynccnt 0" ::: "memory");
#endif
#endif
}

__device__ __forceinline__ float sigmoidf_(float x) { return 1.0f / (1.0f + __expf(-x)); }

// ---------------------------------------------------------------------------
// K0: causal depthwise conv (K=4) + SiLU  ->  u  (B,S,D)
// ---------------------------------------------------------------------------
__global__ __launch_bounds__(256) void conv_silu_kernel(
    const float* __restrict__ x, const float* __restrict__ cw,
    const float* __restrict__ cb, float* __restrict__ u)
{
    int tid = blockIdx.x * 256 + threadIdx.x;            // = ((b*SEQ)+s)*DMODEL + d
    int d  = tid & (DMODEL - 1);
    int sb = tid >> 10;
    int s  = sb & (SEQ - 1);
    int b  = sb >> 12;
    float acc = cb[d];
    #pragma unroll
    for (int k = 0; k < DCONV; ++k) {
        int ss = s - (DCONV - 1) + k;
        if (ss >= 0)
            acc = fmaf(x[(((size_t)b * SEQ + ss) << 10) + d], cw[d * DCONV + k], acc);
    }
    u[tid] = acc * sigmoidf_(acc);
}

// ---------------------------------------------------------------------------
// K1: precompute A_exp = exp(-exp(A)) and A_exp^CHUNKL
// ---------------------------------------------------------------------------
__global__ __launch_bounds__(256) void aexp_kernel(
    const float* __restrict__ A, float* __restrict__ aexp, float* __restrict__ a256)
{
    int tid = blockIdx.x * 256 + threadIdx.x;            // DMODEL*DSTATE = 16384
    float e = __expf(A[tid]);
    aexp[tid] = __expf(-e);
    a256[tid] = __expf(-e * (float)CHUNKL);
}

// ---------------------------------------------------------------------------
// K1b: fp32 -> bf16 bulk conversion (float4 -> 4x bf16 per thread)
// ---------------------------------------------------------------------------
__global__ __launch_bounds__(256) void f32_to_bf16_kernel(
    const float* __restrict__ s, __bf16* __restrict__ d)
{
    int tid = blockIdx.x * 256 + threadIdx.x;
    float4 v = ((const float4*)s)[tid];
    union { __bf16 h[4]; uint2 u; } o;
    o.h[0] = (__bf16)v.x; o.h[1] = (__bf16)v.y;
    o.h[2] = (__bf16)v.z; o.h[3] = (__bf16)v.w;
    ((uint2*)d)[tid] = o.u;
}

// ---------------------------------------------------------------------------
// K2: B/C projections with fp32 WMMA 16x16x4 (keeps scan inputs full precision)
// ---------------------------------------------------------------------------
__global__ __launch_bounds__(256) void bcproj_kernel(
    const float* __restrict__ u,
    const float* __restrict__ Bw, const float* __restrict__ Bb,
    const float* __restrict__ Cw, const float* __restrict__ Cb,
    float* __restrict__ Bt, float* __restrict__ Ct)
{
    int wave = threadIdx.x >> 5, lane = threadIdx.x & 31;
    int tile = blockIdx.x * 8 + wave;                    // 2048 tiles of 16 rows
    int rowbase = tile * 16;
    int lm   = lane & 15;
    int koff = (lane >= 16) ? 2 : 0;
    v8f accB = {}, accC = {};
    const float* urow = &u[((size_t)(rowbase + lm)) << 10];
    const float* brow = &Bw[lm * DMODEL];
    const float* crow = &Cw[lm * DMODEL];
    for (int kb = 0; kb < DMODEL; kb += 4) {
        v2f a, bb, cc;
        a.x  = urow[kb + koff];     a.y  = urow[kb + koff + 1];
        bb.x = brow[kb + koff];     bb.y = brow[kb + koff + 1];
        cc.x = crow[kb + koff];     cc.y = crow[kb + koff + 1];
        accB = __builtin_amdgcn_wmma_f32_16x16x4_f32(false, a, false, bb, (short)0, accB, false, false);
        accC = __builtin_amdgcn_wmma_f32_16x16x4_f32(false, a, false, cc, (short)0, accC, false, false);
    }
    int rhalf = (lane >= 16) ? 8 : 0;
    #pragma unroll
    for (int i = 0; i < 8; ++i) {
        int row = rowbase + i + rhalf;
        Bt[row * DSTATE + lm] = accB[i] + Bb[lm];
        Ct[row * DSTATE + lm] = accC[i] + Cb[lm];
    }
}

// ---------------------------------------------------------------------------
// K3: scan pass 1 — chunk-local end states with h0 = 0
// ---------------------------------------------------------------------------
__global__ __launch_bounds__(256) void scan1_kernel(
    const float* __restrict__ u, const float* __restrict__ Bt,
    const float* __restrict__ aexp, float* __restrict__ hend)
{
    int tid = blockIdx.x * 256 + threadIdx.x;            // B*NCHUNK*DMODEL = 131072
    int d  = tid & (DMODEL - 1);
    int bc = tid >> 10;                                  // b*NCHUNK + chunk
    int b = bc >> 4, chunk = bc & 15;
    float h[DSTATE], ae[DSTATE];
    #pragma unroll
    for (int n = 0; n < DSTATE; ++n) { h[n] = 0.f; ae[n] = aexp[d * DSTATE + n]; }
    int t0 = chunk * CHUNKL;
    for (int tt = 0; tt < CHUNKL; ++tt) {
        size_t r = (size_t)b * SEQ + (t0 + tt);
        float uu = u[(r << 10) + d];
        const float4* bp = (const float4*)&Bt[r * DSTATE];
        float4 b0 = bp[0], b1 = bp[1], b2 = bp[2], b3 = bp[3];
        float bv[DSTATE] = { b0.x,b0.y,b0.z,b0.w, b1.x,b1.y,b1.z,b1.w,
                             b2.x,b2.y,b2.z,b2.w, b3.x,b3.y,b3.z,b3.w };
        #pragma unroll
        for (int n = 0; n < DSTATE; ++n) h[n] = fmaf(h[n], ae[n], uu * bv[n]);
    }
    size_t o = (((size_t)bc << 10) + d) * DSTATE;
    #pragma unroll
    for (int n = 0; n < DSTATE; ++n) hend[o + n] = h[n];
}

// ---------------------------------------------------------------------------
// K4: scan pass 2 — prefix-combine chunk states
// ---------------------------------------------------------------------------
__global__ __launch_bounds__(256) void scan2_kernel(
    const float* __restrict__ hend, const float* __restrict__ a256,
    float* __restrict__ hstart)
{
    int tid = blockIdx.x * 256 + threadIdx.x;            // B*DMODEL*DSTATE = 131072
    int b = tid >> 14;
    int r = tid & 16383;
    int d = r >> 4, n = r & 15;
    float av = a256[d * DSTATE + n];
    float h = 0.f;
    for (int c = 0; c < NCHUNK; ++c) {
        size_t idx = ((((size_t)b * NCHUNK + c) << 10) + d) * DSTATE + n;
        hstart[idx] = h;
        h = fmaf(h, av, hend[idx]);
    }
}

// ---------------------------------------------------------------------------
// K5: scan pass 3 — replay chunk with correct h0, emit y = h·C + D*u
// ---------------------------------------------------------------------------
__global__ __launch_bounds__(256) void scan3_kernel(
    const float* __restrict__ u, const float* __restrict__ Bt,
    const float* __restrict__ Ct, const float* __restrict__ aexp,
    const float* __restrict__ hstart, const float* __restrict__ Dv,
    float* __restrict__ y)
{
    int tid = blockIdx.x * 256 + threadIdx.x;
    int d  = tid & (DMODEL - 1);
    int bc = tid >> 10;
    int b = bc >> 4, chunk = bc & 15;
    float h[DSTATE], ae[DSTATE];
    size_t ho = (((size_t)bc << 10) + d) * DSTATE;
    #pragma unroll
    for (int n = 0; n < DSTATE; ++n) { h[n] = hstart[ho + n]; ae[n] = aexp[d * DSTATE + n]; }
    float Dd = Dv[d];
    int t0 = chunk * CHUNKL;
    for (int tt = 0; tt < CHUNKL; ++tt) {
        size_t r = (size_t)b * SEQ + (t0 + tt);
        float uu = u[(r << 10) + d];
        const float4* bp = (const float4*)&Bt[r * DSTATE];
        const float4* cp = (const float4*)&Ct[r * DSTATE];
        float4 b0 = bp[0], b1 = bp[1], b2 = bp[2], b3 = bp[3];
        float4 c0 = cp[0], c1 = cp[1], c2 = cp[2], c3 = cp[3];
        float bv[DSTATE] = { b0.x,b0.y,b0.z,b0.w, b1.x,b1.y,b1.z,b1.w,
                             b2.x,b2.y,b2.z,b2.w, b3.x,b3.y,b3.z,b3.w };
        float cv[DSTATE] = { c0.x,c0.y,c0.z,c0.w, c1.x,c1.y,c1.z,c1.w,
                             c2.x,c2.y,c2.z,c2.w, c3.x,c3.y,c3.z,c3.w };
        float yo = Dd * uu;
        #pragma unroll
        for (int n = 0; n < DSTATE; ++n) {
            h[n] = fmaf(h[n], ae[n], uu * bv[n]);
            yo   = fmaf(h[n], cv[n], yo);
        }
        y[(r << 10) + d] = yo;
    }
}

// ---------------------------------------------------------------------------
// K6/K7: 128x128-tile GEMM out = Ab @ Wb^T, bf16 WMMA (fp32 accumulate).
//        bf16 operands in global memory; LDS tiles are verbatim 16B copies,
//        staged with async global->LDS (double-buffered) when available.
//        Epilogue: mode 0: outb = bf16(extra * silu(acc + bias))   (gate)
//                  mode 1: outf = acc + bias + extra     (out-proj + residual)
// ---------------------------------------------------------------------------
union FragU { struct { uint4 lo, hi; } q; v16bf v; };

__global__ __launch_bounds__(256) void gemm_bf16_kernel(
    const __bf16* __restrict__ Ab, const __bf16* __restrict__ Wb,
    const float* __restrict__ bias, const float* __restrict__ extra,
    float* __restrict__ outf, __bf16* __restrict__ outb, int mode)
{
    __shared__ __align__(16) __bf16 At[2][128][32];     // [buf][m][k]
    __shared__ __align__(16) __bf16 Wt[2][128][32];     // [buf][n][k]
    const int rowbase = blockIdx.x * 128;
    const int colbase = blockIdx.y * 128;
    const int t    = threadIdx.x;
    const int wave = t >> 5, lane = t & 31;
    const int mrow = (wave & 3) * 32;          // 4 wave-rows of 32
    const int ncol = (wave >> 2) * 64;         // 2 wave-cols of 64
    const int lhalf = lane >> 4;               // 0 or 1
    const int lm    = lane & 15;

    v8f acc[2][4] = {};

    auto stage = [&](int kb, int buf) {
#if USE_ASYNC
        #pragma unroll
        for (int i = 0; i < 2; ++i) {
            int fi = t + i * 256;              // 512 16B chunks per tile
            int r = fi >> 2, q = (fi & 3) << 3;
            async_ld16(&Ab[(size_t)(rowbase + r) * DMODEL + kb + q], &At[buf][r][q]);
            async_ld16(&Wb[(size_t)(colbase + r) * DMODEL + kb + q], &Wt[buf][r][q]);
        }
#else
        uint4 ra[2], rw[2];
        #pragma unroll
        for (int i = 0; i < 2; ++i) {
            int fi = t + i * 256;
            int r = fi >> 2, q = (fi & 3) << 3;
            ra[i] = *(const uint4*)&Ab[(size_t)(rowbase + r) * DMODEL + kb + q];
            rw[i] = *(const uint4*)&Wb[(size_t)(colbase + r) * DMODEL + kb + q];
        }
        #pragma unroll
        for (int i = 0; i < 2; ++i) {
            int fi = t + i * 256;
            int r = fi >> 2, q = (fi & 3) << 3;
            *(uint4*)&At[buf][r][q] = ra[i];
            *(uint4*)&Wt[buf][r][q] = rw[i];
        }
#endif
    };

    stage(0, 0);
    wait_async0();
    __syncthreads();

    int p = 0;
    for (int kb = 0; kb < DMODEL; kb += 32) {
        if (kb + 32 < DMODEL) stage(kb + 32, p ^ 1);

        // fragments from buffer p: each is exactly two 16B LDS reads
        v16bf afrag[2], bfrag[4];
        #pragma unroll
        for (int mt = 0; mt < 2; ++mt) {
            int m = mrow + mt * 16 + lm;
            FragU f;
            f.q.lo = *(const uint4*)&At[p][m][lhalf * 8];        // K in [8h, 8h+8)
            f.q.hi = *(const uint4*)&At[p][m][16 + lhalf * 8];   // K in [16+8h, ...)
            afrag[mt] = f.v;
        }
        #pragma unroll
        for (int nt = 0; nt < 4; ++nt) {
            int n = ncol + nt * 16 + lm;
            FragU f;
            f.q.lo = *(const uint4*)&Wt[p][n][lhalf * 8];        // K in [8h, 8h+8)
            f.q.hi = *(const uint4*)&Wt[p][n][16 + lhalf * 8];
            bfrag[nt] = f.v;
        }
        #pragma unroll
        for (int mt = 0; mt < 2; ++mt)
            #pragma unroll
            for (int nt = 0; nt < 4; ++nt)
                acc[mt][nt] = __builtin_amdgcn_wmma_f32_16x16x32_bf16(
                    false, afrag[mt], false, bfrag[nt], (short)0, acc[mt][nt], false, false);

        wait_async0();
        __syncthreads();
        p ^= 1;
    }

    #pragma unroll
    for (int mt = 0; mt < 2; ++mt) {
        #pragma unroll
        for (int i = 0; i < 8; ++i) {
            int row = rowbase + mrow + mt * 16 + i + lhalf * 8;
            #pragma unroll
            for (int nt = 0; nt < 4; ++nt) {
                int col = colbase + ncol + nt * 16 + lm;
                float v = acc[mt][nt][i] + bias[col];
                size_t idx = ((size_t)row << 10) + col;
                if (mode == 0) outb[idx] = (__bf16)(extra[idx] * (v * sigmoidf_(v)));
                else           outf[idx] = v + extra[idx];
            }
        }
    }
}

// ---------------------------------------------------------------------------
// K8: LayerNorm, one wave per row of 1024
// ---------------------------------------------------------------------------
__global__ __launch_bounds__(256) void ln_kernel(
    const float* __restrict__ z, const float* __restrict__ g,
    const float* __restrict__ bvec, float* __restrict__ out)
{
    int wave = threadIdx.x >> 5, lane = threadIdx.x & 31;
    int row = blockIdx.x * 8 + wave;
    const float* zr = &z[(size_t)row << 10];
    float vals[32];
    float s = 0.f, ss = 0.f;
    #pragma unroll
    for (int i = 0; i < 32; ++i) {
        float v = zr[lane + i * 32];
        vals[i] = v; s += v; ss += v * v;
    }
    #pragma unroll
    for (int off = 16; off >= 1; off >>= 1) {
        s  += __shfl_xor(s,  off, 32);
        ss += __shfl_xor(ss, off, 32);
    }
    float mu  = s * (1.f / DMODEL);
    float var = ss * (1.f / DMODEL) - mu * mu;
    float rin = rsqrtf(var + 1e-5f);
    #pragma unroll
    for (int i = 0; i < 32; ++i) {
        int c = lane + i * 32;
        out[((size_t)row << 10) + c] = (vals[i] - mu) * rin * g[c] + bvec[c];
    }
}

// ---------------------------------------------------------------------------
extern "C" void kernel_launch(void* const* d_in, const int* in_sizes, int n_in,
                              void* d_out, int out_size, void* d_ws, size_t ws_size,
                              hipStream_t stream)
{
    const float* x      = (const float*)d_in[0];
    const float* conv_w = (const float*)d_in[1];
    const float* conv_b = (const float*)d_in[2];
    const float* A      = (const float*)d_in[3];
    const float* Bp_w   = (const float*)d_in[4];
    const float* Bp_b   = (const float*)d_in[5];
    const float* Cp_w   = (const float*)d_in[6];
    const float* Cp_b   = (const float*)d_in[7];
    const float* Dv     = (const float*)d_in[8];
    const float* gate_w = (const float*)d_in[9];
    const float* gate_b = (const float*)d_in[10];
    const float* out_w  = (const float*)d_in[11];
    const float* out_b  = (const float*)d_in[12];
    const float* ln_g   = (const float*)d_in[13];
    const float* ln_b   = (const float*)d_in[14];
    float* outp = (float*)d_out;

    float* wsf = (float*)d_ws;
    const size_t NU = (size_t)ROWS * DMODEL;                 // 33554432
    float*  u      = wsf;  wsf += NU;                        // u, reused as z
    float*  y      = wsf;  wsf += NU;                        // fp32 scan output
    float*  Bt     = wsf;  wsf += (size_t)ROWS * DSTATE;
    float*  Ct     = wsf;  wsf += (size_t)ROWS * DSTATE;
    float*  hend   = wsf;  wsf += (size_t)BATCH * NCHUNK * DMODEL * DSTATE;
    float*  hstart = wsf;  wsf += (size_t)BATCH * NCHUNK * DMODEL * DSTATE;
    float*  aexp   = wsf;  wsf += (size_t)DMODEL * DSTATE;
    float*  a256   = wsf;  wsf += (size_t)DMODEL * DSTATE;
    __bf16* xb     = (__bf16*)wsf;  wsf += NU / 2;           // bf16 copies
    __bf16* ygb    = (__bf16*)wsf;  wsf += NU / 2;
    __bf16* wgb    = (__bf16*)wsf;  wsf += (size_t)DMODEL * DMODEL / 2;
    __bf16* owb    = (__bf16*)wsf;  wsf += (size_t)DMODEL * DMODEL / 2;
    float*  z      = u;                                      // reuse u buffer

    // K0: conv + silu
    conv_silu_kernel<<<(int)(NU / 256), 256, 0, stream>>>(x, conv_w, conv_b, u);
    // K1: A_exp, A_exp^256
    aexp_kernel<<<(DMODEL * DSTATE) / 256, 256, 0, stream>>>(A, aexp, a256);
    // K1b: bf16 copies of GEMM operands
    f32_to_bf16_kernel<<<(int)(NU / 4 / 256), 256, 0, stream>>>(x, xb);
    f32_to_bf16_kernel<<<(DMODEL * DMODEL) / 4 / 256, 256, 0, stream>>>(gate_w, wgb);
    f32_to_bf16_kernel<<<(DMODEL * DMODEL) / 4 / 256, 256, 0, stream>>>(out_w, owb);
    // K2: B/C projections (WMMA f32 16x16x4)
    bcproj_kernel<<<ROWS / 16 / 8, 256, 0, stream>>>(u, Bp_w, Bp_b, Cp_w, Cp_b, Bt, Ct);
    // K3-K5: chunked scan
    scan1_kernel<<<(BATCH * NCHUNK * DMODEL) / 256, 256, 0, stream>>>(u, Bt, aexp, hend);
    scan2_kernel<<<(BATCH * DMODEL * DSTATE) / 256, 256, 0, stream>>>(hend, a256, hstart);
    scan3_kernel<<<(BATCH * NCHUNK * DMODEL) / 256, 256, 0, stream>>>(u, Bt, Ct, aexp, hstart, Dv, y);
    // K6: gate GEMM, ygb <- bf16( y * silu(x @ gate_w^T + gate_b) )
    {
        dim3 grid(ROWS / 128, DMODEL / 128);
        gemm_bf16_kernel<<<grid, 256, 0, stream>>>(xb, wgb, gate_b, y, z, ygb, 0);
    }
    // K7: out GEMM + residual, z <- ygb @ out_w^T + out_b + x
    {
        dim3 grid(ROWS / 128, DMODEL / 128);
        gemm_bf16_kernel<<<grid, 256, 0, stream>>>(ygb, owb, out_b, x, z, ygb, 1);
    }
    // K8: LayerNorm
    ln_kernel<<<ROWS / 8, 256, 0, stream>>>(z, ln_g, ln_b, outp);
}
